// SeqCondAttention_7610682048726
// MI455X (gfx1250) — matrix operand order repaired and checked by reading, not compiled, not run on hardware
//
#include <hip/hip_runtime.h>

typedef __attribute__((ext_vector_type(16))) __bf16 bf16x16;
typedef __attribute__((ext_vector_type(2)))  __bf16 bf16x2;
typedef __attribute__((ext_vector_type(8)))  float  f32x8;

#define DEV __device__ __forceinline__

namespace cfg {
constexpr int B = 4, L = 4096, DMODEL = 768;
constexpr int K = 12, KQ = 6, H = 64, KS = 4;
constexpr int DIM_MEM     = K * H;            // 768
constexpr int DIM_MEM_TOT = DIM_MEM + K;      // 780
constexpr int DC  = DIM_MEM_TOT + KQ * H * 2; // 1548
constexpr int DG  = K * 2 * H;                // 1536
constexpr int DSW = 2 * (H * 3);              // 384
constexpr int DY  = K * DSW / 2;              // 2304
constexpr int CHUNK = 128, NCH = L / CHUNK;   // 32 chunks
constexpr int ROWS = B * L;                   // 16384
}

// ---------------- helpers ----------------
// Native bf16 converts: lets the backend select v_cvt_pk_bf16_f32 instead of
// a 6-op software RNE expansion.
DEV unsigned int pack2bf(float a, float b) {
  union { bf16x2 v; unsigned int u; } x;
  x.v = bf16x2{(__bf16)a, (__bf16)b};
  return x.u;
}
DEV unsigned short f2bf(float a) {
  union { __bf16 h; unsigned short u; } x;
  x.h = (__bf16)a;
  return x.u;
}
DEV float sigmoid_f(float x) { return 1.f / (1.f + __expf(-x)); }
DEV float softplus_f(float x) { return fmaxf(x, 0.f) + log1pf(__expf(-fabsf(x))); }

// depthwise conv (KS=4 causal) + SiLU on one column of z
DEV float conv_silu(const float* __restrict__ zb, const float* __restrict__ cw,
                    int t, int col) {
  const float* zp = zb + col;
  const float* c4 = cw + col * cfg::KS;
  float acc = 0.f;
#pragma unroll
  for (int j = 0; j < cfg::KS; ++j) {
    int tt = t - (cfg::KS - 1) + j;
    float zv = (tt >= 0) ? zp[(size_t)tt * cfg::DC] : 0.f;
    acc = fmaf(zv, c4[j], acc);
  }
  return acc * sigmoid_f(acc);
}

// ================= WMMA GEMM: C[M][N] = A[M][K] * B[N][K]^T =================
// A, B are K-contiguous. EPI: 0 = plain f32 store, 1 = C = C*silu(val) in place.
// ABF16: A is already bf16 (ushort bits) in memory.
// Register-staged double buffering: one barrier per K-step, global loads for
// tile k+1 overlap the WMMAs of tile k.
template <int EPI, bool ABF16>
__global__ __launch_bounds__(256) void gemm_wmma_nt(
    const void* __restrict__ Aptr, const float* __restrict__ Bptr,
    float* __restrict__ Cptr, int N, int Kd) {
  constexpr int LP = 34;          // LDS pitch in bf16 (17 dwords: conflict-free)
  constexpr int BUF = 128 * LP;   // one tile buffer
  __shared__ unsigned short As[2 * BUF];
  __shared__ unsigned short Bs[2 * BUF];

  const int tid = threadIdx.x;
  const int lane = tid & 31, wid = tid >> 5;
  const int wm = wid & 3, wn = wid >> 2;       // 4x2 wave grid
  const int half = lane >> 4, l16 = lane & 15;
  const int bm = blockIdx.y * 128, bn = blockIdx.x * 128;
  const int lr = tid >> 1;                     // loader row 0..127
  const int lc = (tid & 1) << 4;               // loader col 0/16

  const f32x8 zf = {0.f, 0.f, 0.f, 0.f, 0.f, 0.f, 0.f, 0.f};
  f32x8 acc[2][4];
#pragma unroll
  for (int i = 0; i < 2; ++i)
#pragma unroll
    for (int j = 0; j < 4; ++j) acc[i][j] = zf;

  const float* Af32 = (const float*)Aptr;
  const unsigned int* Abf = (const unsigned int*)Aptr;

  unsigned int ra[8], rb[8];                   // staged tile (bf16 pairs)
  auto load_regs = [&](int k0) {
    if (ABF16) {
      const unsigned int* src = Abf + ((size_t)(bm + lr) * Kd + k0 + lc) / 2;
#pragma unroll
      for (int i = 0; i < 8; ++i) ra[i] = src[i];
      if (k0 + 32 < Kd) __builtin_prefetch(src + 16, 0, 0);
    } else {
      const float2* src = (const float2*)(Af32 + (size_t)(bm + lr) * Kd + k0 + lc);
#pragma unroll
      for (int i = 0; i < 8; ++i) { float2 p = src[i]; ra[i] = pack2bf(p.x, p.y); }
      if (k0 + 32 < Kd) __builtin_prefetch(src + 16, 0, 0);
    }
    int nrow = bn + lr;
    if (nrow < N) {
      const float2* src = (const float2*)(Bptr + (size_t)nrow * Kd + k0 + lc);
#pragma unroll
      for (int i = 0; i < 8; ++i) { float2 p = src[i]; rb[i] = pack2bf(p.x, p.y); }
      if (k0 + 32 < Kd) __builtin_prefetch(src + 16, 0, 0);
    } else {
#pragma unroll
      for (int i = 0; i < 8; ++i) rb[i] = 0u;
    }
  };

  load_regs(0);
  int p = 0;
  for (int k0 = 0; k0 < Kd; k0 += 32, p ^= 1) {
    {  // commit staged regs to LDS buffer p
      unsigned int* dA = (unsigned int*)(As + p * BUF + lr * LP + lc);
      unsigned int* dB = (unsigned int*)(Bs + p * BUF + lr * LP + lc);
#pragma unroll
      for (int i = 0; i < 8; ++i) dA[i] = ra[i];
#pragma unroll
      for (int i = 0; i < 8; ++i) dB[i] = rb[i];
    }
    __syncthreads();
    if (k0 + 32 < Kd) load_regs(k0 + 32);  // overlap next-tile loads with WMMAs

    union Frag { bf16x16 v; unsigned int u[8]; };
    Frag fa[2], fb[4];
#pragma unroll
    for (int sm = 0; sm < 2; ++sm) {  // A frag: 16x32, ISA lane layout
      const unsigned int* rp =
          (const unsigned int*)(As + p * BUF + (wm * 32 + sm * 16 + l16) * LP);
#pragma unroll
      for (int v = 0; v < 8; ++v) {
        int kk = ((v >> 2) << 4) | (half << 3) | ((v & 3) << 1);
        fa[sm].u[v] = rp[kk >> 1];
      }
    }
#pragma unroll
    for (int sn = 0; sn < 4; ++sn) {  // B frag: 32x16, ISA lane layout
      const unsigned int* rp =
          (const unsigned int*)(Bs + p * BUF + (wn * 64 + sn * 16 + l16) * LP);
#pragma unroll
      for (int v = 0; v < 8; ++v) {
        int kk = ((v >> 2) << 4) | (half << 3) | ((v & 3) << 1);
        fb[sn].u[v] = rp[kk >> 1];
      }
    }
#pragma unroll
    for (int sm = 0; sm < 2; ++sm)
#pragma unroll
      for (int sn = 0; sn < 4; ++sn)
        acc[sm][sn] = __builtin_amdgcn_wmma_f32_16x16x32_bf16(
            false, fa[sm].v, false, fb[sn].v, (short)0, acc[sm][sn], false, false);
  }

  // epilogue (C/D layout: VGPR r -> M = half*8 + r, N = lane%16)
#pragma unroll
  for (int sm = 0; sm < 2; ++sm)
#pragma unroll
    for (int r = 0; r < 8; ++r) {
      int row = bm + wm * 32 + sm * 16 + half * 8 + r;
#pragma unroll
      for (int sn = 0; sn < 4; ++sn) {
        int col = bn + wn * 64 + sn * 16 + l16;
        if (col < N) {
          float v = acc[sm][sn][r];
          size_t idx = (size_t)row * N + col;
          if (EPI == 1) {
            float o = Cptr[idx];
            Cptr[idx] = o * (v * sigmoid_f(v));  // out_c * silu(gate)
          } else {
            Cptr[idx] = v;
          }
        }
      }
    }
}

// ================= chunked scan (cumsum of p_w, re, im over L) =================
__global__ __launch_bounds__(64) void scan_pass1(
    const float* __restrict__ z, const float* __restrict__ cw,
    const float* __restrict__ ssc, const float* __restrict__ sbi,
    const float* __restrict__ psc, const float* __restrict__ theta_raw,
    const float* __restrict__ decay, float* __restrict__ pden,
    float* __restrict__ pre, float* __restrict__ pim) {
  using namespace cfg;
  int c = blockIdx.x % NCH;
  int bk = blockIdx.x / NCH;
  int k = bk % K, b = bk / K;
  int h = threadIdx.x;
  const float* zb = z + (size_t)b * L * DC;
  float slope = softplus_f(decay[k]);
  float theta = 0.001f + 2.999f * sigmoid_f(theta_raw[k * H + h]);
  float ps = psc[k], sc = ssc[k], sb = sbi[k];
  float den = 0.f, reS = 0.f, imS = 0.f;
  int t0 = c * CHUNK;
  for (int t = t0; t < t0 + CHUNK; ++t) {
    float sraw = conv_silu(zb, cw, t, DIM_MEM + k);
    float pw = softplus_f(fmaf(sc, sraw, sb)) * __expf(-(float)(L - 1 - t) * slope);
    pw = fminf(fmaxf(pw, 1e-6f), 1000.f);
    float kv = conv_silu(zb, cw, t, k * H + h);
    float ksl = kv * ps;
    float phi = ksl / (1.f + fabsf(ksl)) * theta;
    float sp, cp; __sincosf(phi, &sp, &cp);
    float kvw = kv * pw;
    den += pw; reS = fmaf(kvw, cp, reS); imS = fmaf(kvw, sp, imS);
  }
  size_t pi = ((size_t)bk * NCH + c) * H + h;
  pre[pi] = reS; pim[pi] = imS;
  if (h == 0) pden[(size_t)bk * NCH + c] = den;
}

__global__ __launch_bounds__(64) void scan_pass2(float* pden, float* pre, float* pim) {
  using namespace cfg;
  int bk = blockIdx.x, h = threadIdx.x;
  float aR = 0.f, aI = 0.f, aD = 0.f;
  for (int c = 0; c < NCH; ++c) {
    size_t pi = ((size_t)bk * NCH + c) * H + h;
    float r = pre[pi]; pre[pi] = aR; aR += r;
    float i = pim[pi]; pim[pi] = aI; aI += i;
    if (h == 0) { size_t di = (size_t)bk * NCH + c; float d = pden[di]; pden[di] = aD; aD += d; }
  }
}

__global__ __launch_bounds__(64) void scan_pass3(
    const float* __restrict__ z, const float* __restrict__ cw,
    const float* __restrict__ ssc, const float* __restrict__ sbi,
    const float* __restrict__ psc, const float* __restrict__ theta_raw,
    const float* __restrict__ decay, const float* __restrict__ w_int_raw,
    const float* __restrict__ pden, const float* __restrict__ pre,
    const float* __restrict__ pim, float* __restrict__ outc) {
  using namespace cfg;
  int c = blockIdx.x % NCH;
  int bk = blockIdx.x / NCH;
  int k = bk % K, b = bk / K;
  int kq = k >> 1;
  int h = threadIdx.x;
  const float* zb = z + (size_t)b * L * DC;
  float slope = softplus_f(decay[k]);
  float theta = 0.001f + 2.999f * sigmoid_f(theta_raw[k * H + h]);
  float ps = psc[k], sc = ssc[k], sb = sbi[k];
  float ew = __expf(w_int_raw[k * H + h]);
  float wint = ew / (ew + 1e-6f);          // M==1 softmax normalizer
  float den = pden[(size_t)bk * NCH + c];
  size_t pi = ((size_t)bk * NCH + c) * H + h;
  float reS = pre[pi], imS = pim[pi];
  const float rscale = 0.125f;             // 1/sqrt(H)
  int qcol = DIM_MEM_TOT + kq * (2 * H) + 2 * h;
  int t0 = c * CHUNK;
  for (int t = t0; t < t0 + CHUNK; ++t) {
    float sraw = conv_silu(zb, cw, t, DIM_MEM + k);
    float pw = softplus_f(fmaf(sc, sraw, sb)) * __expf(-(float)(L - 1 - t) * slope);
    pw = fminf(fmaxf(pw, 1e-6f), 1000.f);
    float kv = conv_silu(zb, cw, t, k * H + h);
    float ksl = kv * ps;
    float phi = ksl / (1.f + fabsf(ksl)) * theta;
    float sp, cp; __sincosf(phi, &sp, &cp);
    float kvw = kv * pw;
    den += pw; reS = fmaf(kvw, cp, reS); imS = fmaf(kvw, sp, imS);
    float inv = 1.f / fmaxf(den, 1e-4f);
    float s_re = reS * inv, s_im = imS * inv;
    float q_re = conv_silu(zb, cw, t, qcol);
    float q_im = conv_silu(zb, cw, t, qcol + 1);
    float m_re = (s_re * q_re + s_im * q_im) * rscale;
    float m_im = (s_im * q_re - s_re * q_im) * rscale;
    float* op = outc + ((size_t)b * L + t) * DG + k * (2 * H);
    op[h]     = m_re * wint;
    op[H + h] = m_im * wint;
  }
}

// ================= RMS sumsq per row (deterministic, no atomics) =================
__global__ __launch_bounds__(256) void rowsq_reduce(const float* __restrict__ g,
                                                    float* __restrict__ rowsq) {
  __shared__ float sm[256];
  int row = blockIdx.x;
  const float* gr = g + (size_t)row * cfg::DG;
  float s = 0.f;
  for (int c = threadIdx.x; c < cfg::DG; c += 256) { float v = gr[c]; s += v * v; }
  sm[threadIdx.x] = s;
  __syncthreads();
  for (int st = 128; st > 0; st >>= 1) {
    if (threadIdx.x < st) sm[threadIdx.x] += sm[threadIdx.x + st];
    __syncthreads();
  }
  if (threadIdx.x == 0) rowsq[row] = sm[0];
}

// ===== readout: per-k (16384x128)@(128x384) WMMA + SwiGLU, bf16 y_act out =====
__global__ __launch_bounds__(128) void readout_swiglu(
    const float* __restrict__ g, const float* __restrict__ rowsq,
    const float* __restrict__ nw, const float* __restrict__ Wr,
    unsigned short* __restrict__ yact) {
  using namespace cfg;
  int w = threadIdx.x >> 5, lane = threadIdx.x & 31;
  int half = lane >> 4, nl = lane & 15;
  int m0 = blockIdx.x * 64 + w * 16;
  int k = blockIdx.y, np = blockIdx.z;    // np: 16-col pair tile (0..11)

  int arow = m0 + nl;                     // A frag row (M = lane%16)
  float rs = rsqrtf(rowsq[arow] * (1.f / (float)DG) + 1e-6f);
  const float* grow = g + (size_t)arow * DG + k * 128;
  const float* nwk = nw + k * 128;
  const float* WrB = Wr + (size_t)k * 128 * DSW;

  const f32x8 zf = {0.f, 0.f, 0.f, 0.f, 0.f, 0.f, 0.f, 0.f};
  f32x8 accV = zf, accG = zf;
  int colv = np * 16 + nl;

#pragma unroll
  for (int ks = 0; ks < 4; ++ks) {        // K = 128 = 4 x 32
    union Frag { bf16x16 v; unsigned int u[8]; };
    Frag fa, fbv, fbg;
#pragma unroll
    for (int v = 0; v < 8; ++v) {         // A frag: fold rsqrt * norm_weight
      int kk = ((v >> 2) << 4) | (half << 3) | ((v & 3) << 1);
      int f0 = ks * 32 + kk;
      fa.u[v] = pack2bf(grow[f0] * rs * nwk[f0], grow[f0 + 1] * rs * nwk[f0 + 1]);
    }
#pragma unroll
    for (int v = 0; v < 8; ++v) {         // B frags: val cols n, gate cols 192+n
      int kd = ks * 32 + (((v >> 2) << 4) | (half << 3) | ((v & 3) << 1));
      const float* p = WrB + (size_t)kd * DSW + colv;
      fbv.u[v] = pack2bf(p[0], p[DSW]);
      fbg.u[v] = pack2bf(p[192], p[192 + DSW]);
    }
    accV = __builtin_amdgcn_wmma_f32_16x16x32_bf16(false, fa.v, false, fbv.v,
                                                   (short)0, accV, false, false);
    accG = __builtin_amdgcn_wmma_f32_16x16x32_bf16(false, fa.v, false, fbg.v,
                                                   (short)0, accG, false, false);
  }
#pragma unroll
  for (int r = 0; r < 8; ++r) {
    int row = m0 + half * 8 + r;
    float ya = accV[r] * sigmoid_f(accG[r]);  // SwiGLU
    yact[(size_t)row * DY + k * 192 + np * 16 + nl] = f2bf(ya);
  }
}

// ================= host =================
extern "C" void kernel_launch(void* const* d_in, const int* in_sizes, int n_in,
                              void* d_out, int out_size, void* d_ws, size_t ws_size,
                              hipStream_t stream) {
  using namespace cfg;
  (void)in_sizes; (void)n_in; (void)out_size; (void)ws_size;
  const float* x         = (const float*)d_in[0];
  const float* W_in      = (const float*)d_in[1];
  const float* conv_w    = (const float*)d_in[2];
  const float* theta_raw = (const float*)d_in[3];
  const float* w_int_raw = (const float*)d_in[4];
  const float* decay     = (const float*)d_in[5];
  const float* ssc       = (const float*)d_in[6];
  const float* sbi       = (const float*)d_in[7];
  const float* psc       = (const float*)d_in[8];
  const float* W_gate    = (const float*)d_in[9];
  const float* nw        = (const float*)d_in[10];
  const float* Wr        = (const float*)d_in[11];
  const float* W_out     = (const float*)d_in[12];
  float* out = (float*)d_out;

  // workspace layout (~203 MB)
  float* z        = (float*)d_ws;                       // ROWS*DC f32
  float* outc     = z + (size_t)ROWS * DC;              // ROWS*DG f32
  float* part_den = outc + (size_t)ROWS * DG;           // B*K*NCH
  float* part_re  = part_den + (size_t)B * K * NCH;     // B*K*NCH*H
  float* part_im  = part_re + (size_t)B * K * NCH * H;
  float* rowsq    = part_im + (size_t)B * K * NCH * H;  // ROWS
  unsigned short* yact = (unsigned short*)z;            // alias z (dead by then)

  dim3 blk256(256);
  // 1) z = x @ W_in^T
  gemm_wmma_nt<0, false><<<dim3((DC + 127) / 128, ROWS / 128), blk256, 0, stream>>>(
      x, W_in, z, DC, DMODEL);
  // 2) chunked scan -> out_c
  scan_pass1<<<B * K * NCH, 64, 0, stream>>>(z, conv_w, ssc, sbi, psc, theta_raw,
                                             decay, part_den, part_re, part_im);
  scan_pass2<<<B * K, 64, 0, stream>>>(part_den, part_re, part_im);
  scan_pass3<<<B * K * NCH, 64, 0, stream>>>(z, conv_w, ssc, sbi, psc, theta_raw,
                                             decay, w_int_raw, part_den, part_re,
                                             part_im, outc);
  // 3) gate GEMM fused with out_c*silu(gate), in place
  gemm_wmma_nt<1, false><<<dim3(DG / 128, ROWS / 128), blk256, 0, stream>>>(
      x, W_gate, outc, DG, DMODEL);
  // 4) RMS norm sumsq
  rowsq_reduce<<<ROWS, 256, 0, stream>>>(outc, rowsq);
  // 5) readout + SwiGLU -> bf16 y_act
  readout_swiglu<<<dim3(ROWS / 64, K, 12), 128, 0, stream>>>(outc, rowsq, nw, Wr, yact);
  // 6) out = y_act @ W_out^T
  gemm_wmma_nt<0, true><<<dim3(DMODEL / 128, ROWS / 128), blk256, 0, stream>>>(
      yact, W_out, out, DMODEL, DY);
}
